// MultiHeadAttention_74088185856527
// MI455X (gfx1250) — matrix operand order repaired and checked by reading, not compiled, not run on hardware
//
#include <hip/hip_runtime.h>
#include <hip/hip_bf16.h>
#include <type_traits>

typedef __attribute__((ext_vector_type(16))) __bf16 v16bf;
typedef __attribute__((ext_vector_type(8)))  __bf16 v8bf;
typedef __attribute__((ext_vector_type(8)))  float  v8f;
typedef __attribute__((ext_vector_type(4)))  int    v4i;

#define DIM  1024
#define SEQ  2048
#define NB   4
#define NH   16
#define MTOT (NB * SEQ)   // 8192 rows

// ---------------------------------------------------------------------------
// gfx1250 async global->LDS copy (ASYNCcnt) with portable fallback
// ---------------------------------------------------------------------------
#if defined(__HIP_DEVICE_COMPILE__) &&                                   \
    __has_builtin(__builtin_amdgcn_global_load_async_to_lds_b128) &&     \
    __has_builtin(__builtin_amdgcn_s_wait_asynccnt)
#define USE_ASYNC_LDS 1
#else
#define USE_ASYNC_LDS 0
#endif

#if USE_ASYNC_LDS
typedef __attribute__((address_space(1))) v4i gv4i_t;   // global 16B unit
typedef __attribute__((address_space(3))) v4i lv4i_t;   // LDS 16B unit
#endif

static __device__ __forceinline__ void async_copy16(const __bf16* g, __bf16* l) {
#if USE_ASYNC_LDS
  __builtin_amdgcn_global_load_async_to_lds_b128((gv4i_t*)g, (lv4i_t*)l, 0, 0);
#else
  *(uint4*)l = *(const uint4*)g;
#endif
}
static __device__ __forceinline__ void wait_async0() {
#if USE_ASYNC_LDS
  __builtin_amdgcn_s_wait_asynccnt(0);
#endif
}
static __device__ __forceinline__ void wait_async4() {
#if USE_ASYNC_LDS
  __builtin_amdgcn_s_wait_asynccnt(4);
#endif
}

static __device__ __forceinline__ v8f wmma_bf16(v16bf a, v16bf b, v8f c) {
  // D(16x16,f32) = A(16x32,bf16) x B(32x16,bf16) + C
  return __builtin_amdgcn_wmma_f32_16x16x32_bf16(false, a, false, b, (short)0, c, false, false);
}

// ---------------------------------------------------------------------------
// Prep 1: fp32 -> bf16 elementwise (activations), 4 elems / thread
// ---------------------------------------------------------------------------
__global__ __launch_bounds__(256) void cvt_f32_bf16_kernel(const float* __restrict__ src,
                                                           __bf16* __restrict__ dst, int n4) {
  int i = blockIdx.x * 256 + threadIdx.x;
  if (i >= n4) return;
  const float4 v = ((const float4*)src)[i];
  union { __bf16 e[4]; uint2 u; } p;
  p.e[0] = (__bf16)v.x; p.e[1] = (__bf16)v.y; p.e[2] = (__bf16)v.z; p.e[3] = (__bf16)v.w;
  ((uint2*)dst)[i] = p.u;
}

// ---------------------------------------------------------------------------
// Prep 2: W[K][N] fp32 -> WT[N][K] bf16 (32x32 tiles through LDS)
// ---------------------------------------------------------------------------
__global__ __launch_bounds__(256) void transpose_w_kernel(const float* __restrict__ W,
                                                          __bf16* __restrict__ WT) {
  __shared__ __bf16 tile[32][40];   // padded row (80B) keeps uint2 stores aligned
  const int t  = threadIdx.x;
  const int r  = t >> 3;            // 0..31
  const int c4 = (t & 7) << 2;      // 0..28
  const int bk = blockIdx.x * 32, bn = blockIdx.y * 32;
  const float4 v = *(const float4*)(W + (size_t)(bk + r) * DIM + bn + c4);
  tile[c4 + 0][r] = (__bf16)v.x;
  tile[c4 + 1][r] = (__bf16)v.y;
  tile[c4 + 2][r] = (__bf16)v.z;
  tile[c4 + 3][r] = (__bf16)v.w;
  __syncthreads();
  union { __bf16 e[4]; uint2 u; } p;
  p.e[0] = tile[r][c4 + 0]; p.e[1] = tile[r][c4 + 1];
  p.e[2] = tile[r][c4 + 2]; p.e[3] = tile[r][c4 + 3];
  *(uint2*)(WT + (size_t)(bn + r) * DIM + bk + c4) = p.u;
}

// ---------------------------------------------------------------------------
// WMMA GEMM: C[M,N] = A[M,K] * BT[N,K]^T   (A, BT bf16 row-major)
// 128x128 block tile, 8 waves (2x4), 64x32 per wave (4x2 fragments).
// Double-buffered LDS filled with async global->LDS; next tile is staged while
// WMMAs consume the current one (wait asynccnt<=4 keeps one stage in flight).
// ---------------------------------------------------------------------------
template <typename OT>
__global__ __launch_bounds__(256) void wmma_gemm_kernel(const __bf16* __restrict__ A,
                                                        const __bf16* __restrict__ BT,
                                                        OT* __restrict__ C,
                                                        int N, int K) {
  __shared__ __bf16 ldsA[2][128 * 32];   // [row][k]
  __shared__ __bf16 ldsB[2][128 * 32];   // [col][k]

  const int tid  = threadIdx.x;
  const int w    = tid >> 5;
  const int lane = tid & 31;
  const int m16  = lane & 15;
  const int half = lane >> 4;
  const int wrow = (w >> 2) * 64;
  const int wcol = (w & 3) * 32;
  const int bm = blockIdx.x, bn = blockIdx.y;

  auto stage = [&](int buf, int kt) {
#pragma unroll
    for (int t = 0; t < 2; ++t) {
      int idx = tid + t * 256;          // 512 x 8-element units
      int row = idx >> 2;               // 0..127
      int c8  = (idx & 3) << 3;         // 0..24
      async_copy16(A  + (size_t)(bm * 128 + row) * K + kt + c8, &ldsA[buf][row * 32 + c8]);
      async_copy16(BT + (size_t)(bn * 128 + row) * K + kt + c8, &ldsB[buf][row * 32 + c8]);
    }
  };

  v8f acc[4][2] = {};
  const int nk = K >> 5;
  stage(0, 0);
  for (int kt = 0; kt < nk; ++kt) {
    const int cur = kt & 1;
    if (kt + 1 < nk) { stage(cur ^ 1, (kt + 1) << 5); wait_async4(); }
    else             { wait_async0(); }
    __syncthreads();

    v16bf a[4], b[2];
#pragma unroll
    for (int i = 0; i < 4; ++i) {
      const __bf16* p = &ldsA[cur][(wrow + i * 16 + m16) * 32];
      ((v8bf*)&a[i])[0] = *(const v8bf*)(p + half * 8);       // e0..7 : k = e + 8*half
      ((v8bf*)&a[i])[1] = *(const v8bf*)(p + 16 + half * 8);  // e8..15: k = 16+(e-8)+8*half
    }
#pragma unroll
    for (int j = 0; j < 2; ++j) {
      const __bf16* p = &ldsB[cur][(wcol + j * 16 + m16) * 32];  // same k order as A
      ((v8bf*)&b[j])[0] = *(const v8bf*)(p + half * 8);
      ((v8bf*)&b[j])[1] = *(const v8bf*)(p + 16 + half * 8);
    }
#pragma unroll
    for (int i = 0; i < 4; ++i)
#pragma unroll
      for (int j = 0; j < 2; ++j)
        acc[i][j] = wmma_bf16(a[i], b[j], acc[i][j]);
    __syncthreads();
  }

#pragma unroll
  for (int i = 0; i < 4; ++i)
#pragma unroll
    for (int j = 0; j < 2; ++j)
#pragma unroll
      for (int r = 0; r < 8; ++r) {
        int row = bm * 128 + wrow + i * 16 + r + 8 * half;
        int col = bn * 128 + wcol + j * 16 + m16;
        float v = acc[i][j][r];
        if constexpr (std::is_same<OT, float>::value)
          C[(size_t)row * N + col] = v;
        else
          C[(size_t)row * N + col] = (__bf16)v;
      }
}

// ---------------------------------------------------------------------------
// Causal flash attention. Grid: (SEQ/64, NB*NH), block: 128 threads = 4 waves.
// Wave owns a 16-row query tile (d=64); 32-key chunks, online softmax,
// scores and P*V on bf16 WMMA with f32 accumulation.
// ---------------------------------------------------------------------------
__global__ __launch_bounds__(128) void flash_attn_kernel(const __bf16* __restrict__ XQ,
                                                         const __bf16* __restrict__ XK,
                                                         const __bf16* __restrict__ XV,
                                                         __bf16* __restrict__ XO) {
  __shared__ __bf16 ldsQ[64 * 64];      // [qrow][d]
  __shared__ __bf16 ldsK[32 * 64];      // [key][d]
  __shared__ __bf16 ldsVT[64 * 32];     // [d][key]
  __shared__ __bf16 ldsP[4][16 * 32];   // per-wave P transpose scratch

  const int tid  = threadIdx.x;
  const int w    = tid >> 5;
  const int lane = tid & 31;
  const int m16  = lane & 15;
  const int half = lane >> 4;
  const int qt = blockIdx.x;
  const int b  = blockIdx.y >> 4;
  const int h  = blockIdx.y & 15;
  const int q0 = qt * 64;
  const int qbase = q0 + w * 16;
  const size_t bbase = (size_t)b * SEQ;
  const int hoff = h * 64;

  { // stage Q tile 64x64 (async, 4 x 16B per thread)
    int row = tid >> 1, cseg = (tid & 1) * 32;
    const __bf16* src = XQ + (bbase + q0 + row) * DIM + hoff + cseg;
    __bf16* dst = &ldsQ[row * 64 + cseg];
#pragma unroll
    for (int jj = 0; jj < 4; ++jj) async_copy16(src + jj * 8, dst + jj * 8);
  }
  wait_async0();
  __syncthreads();

  v16bf qa[2];  // A-fragments, d-contraction split into 2 chunks of 32
#pragma unroll
  for (int c = 0; c < 2; ++c) {
    const __bf16* p = &ldsQ[(w * 16 + m16) * 64 + c * 32];
    ((v8bf*)&qa[c])[0] = *(const v8bf*)(p + half * 8);
    ((v8bf*)&qa[c])[1] = *(const v8bf*)(p + 16 + half * 8);
  }

  v8f O[4] = {};
  float mrow[8], lrow[8];
#pragma unroll
  for (int r = 0; r < 8; ++r) { mrow[r] = -1e30f; lrow[r] = 0.f; }

  for (int k0 = 0; k0 < q0 + 64; k0 += 32) {
    { // stage K chunk [32 keys][64] (async)
      int key = tid >> 2, seg = (tid & 3) << 4;
      const __bf16* src = XK + (bbase + k0 + key) * DIM + hoff + seg;
      __bf16* dst = &ldsK[key * 64 + seg];
      async_copy16(src, dst);
      async_copy16(src + 8, dst + 8);
    }
    { // stage V chunk transposed [64 d][32 keys] (manual: scatter)
      int key = tid & 31, dseg = (tid >> 5) << 4;
      const uint4* src = (const uint4*)(XV + (bbase + k0 + key) * DIM + hoff + dseg);
      union { uint4 u[2]; __bf16 e[16]; } tmp;
      tmp.u[0] = src[0]; tmp.u[1] = src[1];
#pragma unroll
      for (int e = 0; e < 16; ++e) ldsVT[(dseg + e) * 32 + key] = tmp.e[e];
    }
    wait_async0();
    __syncthreads();

    // scores: S(16x32) = Q(16x64) * K^T
    v8f S0 = {}, S1 = {};
#pragma unroll
    for (int c = 0; c < 2; ++c) {
      v16bf kb0, kb1;
      const __bf16* p0 = &ldsK[m16 * 64 + c * 32];
      ((v8bf*)&kb0)[0] = *(const v8bf*)(p0 + half * 8);
      ((v8bf*)&kb0)[1] = *(const v8bf*)(p0 + 16 + half * 8);
      const __bf16* p1 = &ldsK[(16 + m16) * 64 + c * 32];
      ((v8bf*)&kb1)[0] = *(const v8bf*)(p1 + half * 8);
      ((v8bf*)&kb1)[1] = *(const v8bf*)(p1 + 16 + half * 8);
      S0 = wmma_bf16(qa[c], kb0, S0);
      S1 = wmma_bf16(qa[c], kb1, S1);
    }

    // online softmax (row stats replicated across the 16 lanes of each half)
#pragma unroll
    for (int r = 0; r < 8; ++r) {
      const int qrow = qbase + r + 8 * half;
      float x0 = S0[r] * 0.125f;              // 1/sqrt(64)
      float x1 = S1[r] * 0.125f;
      if (k0 + m16 > qrow)      x0 = -1e30f;  // causal mask
      if (k0 + 16 + m16 > qrow) x1 = -1e30f;
      float cur = fmaxf(x0, x1);
#pragma unroll
      for (int off = 8; off >= 1; off >>= 1) cur = fmaxf(cur, __shfl_xor(cur, off, 16));
      const float mnew  = fmaxf(mrow[r], cur);
      const float alpha = __expf(mrow[r] - mnew);
      const float p0 = __expf(x0 - mnew);
      const float p1 = __expf(x1 - mnew);
      float ps = p0 + p1;
#pragma unroll
      for (int off = 8; off >= 1; off >>= 1) ps += __shfl_xor(ps, off, 16);
      lrow[r] = lrow[r] * alpha + ps;
      mrow[r] = mnew;
#pragma unroll
      for (int f = 0; f < 4; ++f) O[f][r] *= alpha;
      ldsP[w][(r + 8 * half) * 32 + m16]      = (__bf16)p0;
      ldsP[w][(r + 8 * half) * 32 + 16 + m16] = (__bf16)p1;
    }
    asm volatile("s_wait_dscnt 0" ::: "memory");  // wave-private LDS RAW

    v16bf pa;
    {
      const __bf16* pp = &ldsP[w][m16 * 32];
      ((v8bf*)&pa)[0] = *(const v8bf*)(pp + half * 8);
      ((v8bf*)&pa)[1] = *(const v8bf*)(pp + 16 + half * 8);
    }
#pragma unroll
    for (int f = 0; f < 4; ++f) {
      v16bf vb;
      const __bf16* vp = &ldsVT[(f * 16 + m16) * 32];
      ((v8bf*)&vb)[0] = *(const v8bf*)(vp + half * 8);
      ((v8bf*)&vb)[1] = *(const v8bf*)(vp + 16 + half * 8);
      O[f] = wmma_bf16(pa, vb, O[f]);
    }
    __syncthreads();
  }

  // normalize + store (XO is the qb buffer, dead after the Q projection;
  // it aliases none of XQ, XK, XV)
#pragma unroll
  for (int f = 0; f < 4; ++f)
#pragma unroll
    for (int r = 0; r < 8; ++r) {
      int row = qbase + r + 8 * half;
      int col = hoff + f * 16 + m16;
      float v = O[f][r] / lrow[r];
      XO[(bbase + row) * DIM + col] = (__bf16)v;
    }
}

// ---------------------------------------------------------------------------
extern "C" void kernel_launch(void* const* d_in, const int* in_sizes, int n_in,
                              void* d_out, int out_size, void* d_ws, size_t ws_size,
                              hipStream_t stream) {
  (void)in_sizes; (void)n_in; (void)out_size; (void)ws_size;
  const float* q  = (const float*)d_in[0];
  const float* k  = (const float*)d_in[1];
  const float* v  = (const float*)d_in[2];
  const float* wq = (const float*)d_in[3];
  const float* wk = (const float*)d_in[4];
  const float* wv = (const float*)d_in[5];
  const float* wo = (const float*)d_in[6];
  float* out = (float*)d_out;

  const size_t X = (size_t)MTOT * DIM;      // activation elements (8.4M)
  const size_t W = (size_t)DIM * DIM;       // weight elements (1M)
  __bf16* qb  = (__bf16*)d_ws;              // bf16 activations
  __bf16* kb  = qb + X;
  __bf16* vb  = kb + X;
  __bf16* xq  = vb + X;                     // projected Q/K/V
  __bf16* xk  = xq + X;
  __bf16* xv  = xk + X;
  __bf16* wtq = xv + X;                     // transposed bf16 weights
  __bf16* wtk = wtq + W;
  __bf16* wtv = wtk + W;
  __bf16* wto = wtv + W;
  __bf16* xo  = qb;                         // attention out (qb is dead by then)

  const int n4 = (int)(X / 4);
  cvt_f32_bf16_kernel<<<n4 / 256, 256, 0, stream>>>(q, qb, n4);
  cvt_f32_bf16_kernel<<<n4 / 256, 256, 0, stream>>>(k, kb, n4);
  cvt_f32_bf16_kernel<<<n4 / 256, 256, 0, stream>>>(v, vb, n4);

  dim3 tg(DIM / 32, DIM / 32);
  transpose_w_kernel<<<tg, 256, 0, stream>>>(wq, wtq);
  transpose_w_kernel<<<tg, 256, 0, stream>>>(wk, wtk);
  transpose_w_kernel<<<tg, 256, 0, stream>>>(wv, wtv);
  transpose_w_kernel<<<tg, 256, 0, stream>>>(wo, wto);

  dim3 gg(MTOT / 128, DIM / 128);           // (64, 8)
  wmma_gemm_kernel<__bf16><<<gg, 256, 0, stream>>>(qb, wtq, xq, DIM, DIM);
  wmma_gemm_kernel<__bf16><<<gg, 256, 0, stream>>>(kb, wtk, xk, DIM, DIM);
  wmma_gemm_kernel<__bf16><<<gg, 256, 0, stream>>>(vb, wtv, xv, DIM, DIM);

  flash_attn_kernel<<<dim3(SEQ / 64, NB * NH), 128, 0, stream>>>(xq, xk, xv, xo);

  wmma_gemm_kernel<float><<<gg, 256, 0, stream>>>(xo, wto, out, DIM, DIM);
}
